// Gemma3Attention_58514634441001
// MI455X (gfx1250) — matrix operand order repaired and checked by reading, not compiled, not run on hardware
//
#include <hip/hip_runtime.h>
#include <hip/hip_bf16.h>

typedef __bf16 bf16_t;
typedef bf16_t v2bf  __attribute__((ext_vector_type(2)));
typedef bf16_t v4bf  __attribute__((ext_vector_type(4)));
typedef bf16_t v8bf  __attribute__((ext_vector_type(8)));
typedef bf16_t v16bf __attribute__((ext_vector_type(16)));
typedef float  v8f   __attribute__((ext_vector_type(8)));
typedef unsigned int u32x4 __attribute__((ext_vector_type(4)));
typedef int          i32x8 __attribute__((ext_vector_type(8)));
typedef int          i32x4 __attribute__((ext_vector_type(4)));

#define B_       2
#define S_       4096
#define DIM_     2048
#define WIN_     1024
#define SCALING_ 0.0625f   // 256^-0.5

static __device__ __forceinline__ v16bf make_frag(const void* lo_p, const void* hi_p) {
  v8bf lo = *(const v8bf*)lo_p;
  v8bf hi = *(const v8bf*)hi_p;
  return __builtin_shufflevector(lo, hi, 0,1,2,3,4,5,6,7,8,9,10,11,12,13,14,15);
}

static __device__ __forceinline__ v8f wmma_bf16(v16bf a, v16bf b, v8f c) {
  // 8 args: (neg_a, A, neg_b, B, c_mod, C, reuse_a, reuse_b)
  return __builtin_amdgcn_wmma_f32_16x16x32_bf16(false, a, false, b, (short)0, c,
                                                 false, false);
}

static __device__ __forceinline__ float redmax16(float v) {
  v = fmaxf(v, __shfl_xor(v, 1, 32));
  v = fmaxf(v, __shfl_xor(v, 2, 32));
  v = fmaxf(v, __shfl_xor(v, 4, 32));
  v = fmaxf(v, __shfl_xor(v, 8, 32));
  return v;
}
static __device__ __forceinline__ float redsum16(float v) {
  v += __shfl_xor(v, 1, 32);
  v += __shfl_xor(v, 2, 32);
  v += __shfl_xor(v, 4, 32);
  v += __shfl_xor(v, 8, 32);
  return v;
}

// ---------------------------------------------------------------------------
// Tensor Data Mover: 2D tile load global->LDS with per-row LDS padding.
//   tile_d0 elems per row, tile_d1 rows, data_size=2B (bf16),
//   global row stride in elements; LDS padding per (pad_interval, pad_amount).
// D# packing per CDNA5 ISA 8.3/8.4 (count=1, type=2, no cluster multicast).
// ---------------------------------------------------------------------------
static __device__ __forceinline__ unsigned lds_offset_of(const void* p) {
  return (unsigned)(unsigned long long)
      (const __attribute__((address_space(3))) unsigned char*)p;
}

static __device__ __forceinline__ void tdm_load_2d_bf16(
    unsigned lds_addr, const void* gaddr, unsigned tile_d0, unsigned tile_d1,
    unsigned long long row_stride_elems, unsigned pad_interval_code,
    unsigned pad_amount_code) {
  const unsigned long long ga = (unsigned long long)gaddr;
  u32x4 g0;
  g0.x = 1u;                                        // count=1, is_restore=0
  g0.y = lds_addr;                                  // LDS byte address
  g0.z = (unsigned)(ga & 0xffffffffull);            // global_addr[31:0]
  g0.w = (unsigned)((ga >> 32) & 0x01ffffffull)     // global_addr[56:32]
         | (2u << 30);                              // type = 2 ("image")
  const unsigned td0 = 0x7fffffffu, td1 = 0x7fffffffu;  // huge dims: no OOB clip
  i32x8 g1;
  g1[0] = (int)((1u << 16)                          // data_size = 2 bytes
                | (1u << 20)                        // pad_enable
                | (pad_interval_code << 22) | (pad_amount_code << 25));
  g1[1] = (int)((td0 & 0xffffu) << 16);             // tensor_dim0[15:0] (abar=0)
  g1[2] = (int)(((td1 & 0xffffu) << 16) | (td0 >> 16));
  g1[3] = (int)((tile_d0 << 16) | (td1 >> 16));
  g1[4] = (int)tile_d1;                             // tile_dim1 (tile_dim2=0)
  g1[5] = (int)(row_stride_elems & 0xffffffffull);  // tensor_dim0_stride lo
  g1[6] = (int)((row_stride_elems >> 32) & 0xffffull);
  g1[7] = 0;
  const i32x4 gz = {0, 0, 0, 0};
#if __clang_major__ >= 23
  const i32x8 gz8 = {0, 0, 0, 0, 0, 0, 0, 0};
  __builtin_amdgcn_tensor_load_to_lds(g0, g1, gz, gz, gz8, 0);
#else
  __builtin_amdgcn_tensor_load_to_lds(g0, g1, gz, gz, 0);
#endif
}

// ---------------------------------------------------------------------------
// GEMM: Y[M,N] (bf16) = X[M,K] (f32) @ W[K,N] (f32), M=8192 via grid.y, 128x128 tiles
// (f32 sources need VALU conversion, so tiles are staged manually here)
// ---------------------------------------------------------------------------
__global__ __launch_bounds__(256) void gemm_f32_bf16(const float* __restrict__ X,
                                                     const float* __restrict__ W,
                                                     bf16_t* __restrict__ Y,
                                                     int N, int Kd) {
  __shared__ __align__(16) unsigned char ldsA[128 * 80]; // [row][k] bf16, 64B +16 pad
  __shared__ __align__(16) unsigned char ldsB[128 * 80]; // [n][k]   bf16 (transposed)
  const int t = threadIdx.x;
  const int lane = t & 31;
  const int wid = t >> 5;
  const int waveM = wid >> 1, waveN = wid & 1;
  const int m0 = blockIdx.y * 128, n0 = blockIdx.x * 128;
  const int ln = lane & 15, lh = lane >> 4;

  v8f acc[2][4];
  const v8f vzero = {0.f,0.f,0.f,0.f,0.f,0.f,0.f,0.f};
#pragma unroll
  for (int i = 0; i < 2; i++)
#pragma unroll
    for (int j = 0; j < 4; j++) acc[i][j] = vzero;

  for (int kt = 0; kt < Kd; kt += 32) {
    __syncthreads();
    {
      const int kc = (t & 7) * 4;
#pragma unroll
      for (int rr = 0; rr < 4; rr++) {
        const int row = (t >> 3) + rr * 32;
        const float4 v = *(const float4*)(X + (size_t)(m0 + row) * Kd + kt + kc);
        v4bf h = {(bf16_t)v.x, (bf16_t)v.y, (bf16_t)v.z, (bf16_t)v.w};
        *(v4bf*)(ldsA + row * 80 + kc * 2) = h;
      }
    }
    {
      const int n = t & 127;
#pragma unroll
      for (int p = 0; p < 8; p++) {
        const int kp = (t >> 7) + p * 2;
        const float* wp = W + (size_t)(kt + kp * 2) * N + n0 + n;
        v2bf h = {(bf16_t)wp[0], (bf16_t)wp[N]};
        *(v2bf*)(ldsB + n * 80 + kp * 4) = h;
      }
    }
    __syncthreads();

    v16bf af[2], bfr[4];
#pragma unroll
    for (int mi = 0; mi < 2; mi++) {
      const int row = waveM * 32 + mi * 16 + ln;
      af[mi] = make_frag(ldsA + row * 80 + lh * 16,
                         ldsA + row * 80 + 32 + lh * 16);
    }
#pragma unroll
    for (int ni = 0; ni < 4; ni++) {
      const int n = waveN * 64 + ni * 16 + ln;
      bfr[ni] = make_frag(ldsB + n * 80 + lh * 16,
                          ldsB + n * 80 + 32 + lh * 16);
    }
#pragma unroll
    for (int mi = 0; mi < 2; mi++)
#pragma unroll
      for (int ni = 0; ni < 4; ni++)
        acc[mi][ni] = wmma_bf16(af[mi], bfr[ni], acc[mi][ni]);
  }

#pragma unroll
  for (int mi = 0; mi < 2; mi++)
#pragma unroll
    for (int ni = 0; ni < 4; ni++) {
      const int n = n0 + waveN * 64 + ni * 16 + ln;
#pragma unroll
      for (int r = 0; r < 8; r++) {
        const int m = m0 + waveM * 32 + mi * 16 + r + 8 * lh;
        Y[(size_t)m * N + n] = (bf16_t)acc[mi][ni][r];
      }
    }
}

// ---------------------------------------------------------------------------
// GEMM: Y[M,N] (f32) = A[M,K] (bf16) @ W[K,N] (f32)  -- out projection
// A tile staged by the Tensor Data Mover (bf16 -> bf16, row pad 64B->80B).
// ---------------------------------------------------------------------------
__global__ __launch_bounds__(256) void gemm_bf16_f32(const bf16_t* __restrict__ A,
                                                     const float* __restrict__ W,
                                                     float* __restrict__ Y,
                                                     int N, int Kd) {
  __shared__ __align__(16) unsigned char ldsA[128 * 80];
  __shared__ __align__(16) unsigned char ldsB[128 * 80];
  const int t = threadIdx.x;
  const int lane = t & 31;
  const int wid = t >> 5;
  const int waveM = wid >> 1, waveN = wid & 1;
  const int m0 = blockIdx.y * 128, n0 = blockIdx.x * 128;
  const int ln = lane & 15, lh = lane >> 4;

  v8f acc[2][4];
  const v8f vzero = {0.f,0.f,0.f,0.f,0.f,0.f,0.f,0.f};
#pragma unroll
  for (int i = 0; i < 2; i++)
#pragma unroll
    for (int j = 0; j < 4; j++) acc[i][j] = vzero;

  for (int kt = 0; kt < Kd; kt += 32) {
    __syncthreads();
    if (wid == 0) {
      // 128 rows x 32 bf16; global row stride Kd elems; LDS 16 DW + 4 DW pad
      tdm_load_2d_bf16(lds_offset_of(ldsA), A + (size_t)m0 * Kd + kt,
                       32u, 128u, (unsigned long long)Kd,
                       /*pad_interval 2^(3+1)=16 DW*/ 3u, /*pad 4 DW*/ 3u);
    }
    {
      const int n = t & 127;
#pragma unroll
      for (int p = 0; p < 8; p++) {
        const int kp = (t >> 7) + p * 2;
        const float* wp = W + (size_t)(kt + kp * 2) * N + n0 + n;
        v2bf h = {(bf16_t)wp[0], (bf16_t)wp[N]};
        *(v2bf*)(ldsB + n * 80 + kp * 4) = h;
      }
    }
    if (wid == 0) __builtin_amdgcn_s_wait_tensorcnt(0);
    __syncthreads();

    v16bf af[2], bfr[4];
#pragma unroll
    for (int mi = 0; mi < 2; mi++) {
      const int row = waveM * 32 + mi * 16 + ln;
      af[mi] = make_frag(ldsA + row * 80 + lh * 16,
                         ldsA + row * 80 + 32 + lh * 16);
    }
#pragma unroll
    for (int ni = 0; ni < 4; ni++) {
      const int n = waveN * 64 + ni * 16 + ln;
      bfr[ni] = make_frag(ldsB + n * 80 + lh * 16,
                          ldsB + n * 80 + 32 + lh * 16);
    }
#pragma unroll
    for (int mi = 0; mi < 2; mi++)
#pragma unroll
      for (int ni = 0; ni < 4; ni++)
        acc[mi][ni] = wmma_bf16(af[mi], bfr[ni], acc[mi][ni]);
  }

#pragma unroll
  for (int mi = 0; mi < 2; mi++)
#pragma unroll
    for (int ni = 0; ni < 4; ni++) {
      const int n = n0 + waveN * 64 + ni * 16 + ln;
#pragma unroll
      for (int r = 0; r < 8; r++) {
        const int m = m0 + waveM * 32 + mi * 16 + r + 8 * lh;
        Y[(size_t)m * N + n] = acc[mi][ni][r];
      }
    }
}

// ---------------------------------------------------------------------------
// Fused per-head RMSNorm + RoPE, in-place on bf16 [B*S, H, 256].
// ---------------------------------------------------------------------------
__global__ __launch_bounds__(256) void rmsrope_kernel(bf16_t* __restrict__ X,
                                                      const float* __restrict__ Wn,
                                                      int H) {
  const int lane = threadIdx.x & 31;
  const int wv = threadIdx.x >> 5;
  const int idx = blockIdx.x * 8 + wv;
  const int token = idx / H;
  const int head = idx - token * H;
  const int s = token & (S_ - 1);
  bf16_t* p = X + (size_t)token * H * 256 + head * 256 + lane * 8;

  v8bf xv = *(const v8bf*)p;
  float x[8];
  float ss = 0.f;
#pragma unroll
  for (int j = 0; j < 8; j++) { x[j] = (float)xv[j]; ss += x[j] * x[j]; }
  ss += __shfl_xor(ss, 1, 32);
  ss += __shfl_xor(ss, 2, 32);
  ss += __shfl_xor(ss, 4, 32);
  ss += __shfl_xor(ss, 8, 32);
  ss += __shfl_xor(ss, 16, 32);
  const float rn = rsqrtf(ss * (1.0f / 256.0f) + 1e-6f);

  const float* wn = Wn + lane * 8;
  v8bf out;
#pragma unroll
  for (int j = 0; j < 4; j++) {
    const float xr = x[2 * j] * rn * wn[2 * j];
    const float xi = x[2 * j + 1] * rn * wn[2 * j + 1];
    const int pidx = lane * 4 + j;                 // rope pair index 0..127
    const float ang = (float)s * __expf(-(float)pidx * 0.0719557843f);
    float sn, cs;
    __sincosf(ang, &sn, &cs);
    out[2 * j] = (bf16_t)(xr * cs - xi * sn);
    out[2 * j + 1] = (bf16_t)(xr * sn + xi * cs);
  }
  *(v8bf*)p = out;
}

// ---------------------------------------------------------------------------
// Flash attention, sliding window 1024, GQA (kv head = h/2).
// Block = 4 waves; 64 q rows per block (16 per wave); 32-key tiles.
// Q and K tiles staged by the Tensor Data Mover (row pad 512B -> 528B).
// ---------------------------------------------------------------------------
__global__ __launch_bounds__(128) void attn_kernel(const bf16_t* __restrict__ Q,
                                                   const bf16_t* __restrict__ K,
                                                   const bf16_t* __restrict__ V,
                                                   bf16_t* __restrict__ O) {
  __shared__ __align__(16) unsigned char ldsQ[64 * 528];   // [qrow][d]  bf16
  __shared__ __align__(16) unsigned char ldsK[32 * 528];   // [key][d]   bf16
  __shared__ __align__(16) unsigned char ldsV[256 * 80];   // [d][key]   bf16 (transposed)
  __shared__ __align__(16) unsigned char ldsP[4 * 16 * 80]; // per-wave P transpose

  const int t = threadIdx.x;
  const int lane = t & 31, w = t >> 5;
  const int ln = lane & 15, lh = lane >> 4;
  const int q0 = blockIdx.x * 64;
  const int h = blockIdx.y, b = blockIdx.z;
  const int kvh = h >> 1;
  const size_t qbase = (size_t)b * S_ * 2048 + (size_t)h * 256;
  const size_t kbase = (size_t)b * S_ * 1024 + (size_t)kvh * 256;

  // ---- stage Q tile (64 x 256 bf16) via TDM; first in-loop wait covers it ----
  if (w == 0) {
    tdm_load_2d_bf16(lds_offset_of(ldsQ), Q + qbase + (size_t)q0 * 2048,
                     256u, 64u, 2048ull,
                     /*pad_interval 2^(6+1)=128 DW*/ 6u, /*pad 4 DW*/ 3u);
  }

  v8f o[16];
  const v8f vzero = {0.f,0.f,0.f,0.f,0.f,0.f,0.f,0.f};
#pragma unroll
  for (int f = 0; f < 16; f++) o[f] = vzero;
  float mi_[8], li_[8];
#pragma unroll
  for (int r = 0; r < 8; r++) { mi_[r] = -3.0e38f; li_[r] = 0.f; }

  int ks = q0 - WIN_ + 1;
  if (ks < 0) ks = 0;
  ks &= ~31;

  for (int kb = ks; kb <= q0 + 63; kb += 32) {
    __syncthreads();
    // ---- K tile: 32 keys x 256 d via TDM ----
    if (w == 0) {
      tdm_load_2d_bf16(lds_offset_of(ldsK), K + kbase + (size_t)kb * 1024,
                       256u, 32u, 1024ull, 6u, 3u);
    }
    // ---- V tile transposed manually: [d][key] pairs along key ----
#pragma unroll
    for (int j = 0; j < 32; j++) {
      const int sid = t + j * 128;
      const int d = sid & 255, kp = sid >> 8;
      const bf16_t* vp = V + kbase + (size_t)(kb + kp * 2) * 1024 + d;
      v2bf pr = {vp[0], vp[1024]};
      *(v2bf*)(ldsV + d * 80 + kp * 4) = pr;
    }
    if (w == 0) __builtin_amdgcn_s_wait_tensorcnt(0);
    __syncthreads();

    // ---- scores S = Q @ K^T (16 rows x 32 keys per wave) ----
    v8f sf[2] = {vzero, vzero};
    const int qrow = w * 16 + ln;
#pragma unroll
    for (int kd = 0; kd < 8; kd++) {
      v16bf aq = make_frag(ldsQ + qrow * 528 + kd * 64 + lh * 16,
                           ldsQ + qrow * 528 + kd * 64 + 32 + lh * 16);
#pragma unroll
      for (int ni = 0; ni < 2; ni++) {
        const int key = ni * 16 + ln;
        v16bf bk = make_frag(ldsK + key * 528 + kd * 64 + lh * 16,
                             ldsK + key * 528 + kd * 64 + 32 + lh * 16);
        sf[ni] = wmma_bf16(aq, bk, sf[ni]);
      }
    }

    // ---- mask + online softmax (row r: VGPR r, row = r + 8*lh) ----
    const int j0 = kb + ln, j1 = kb + 16 + ln;
#pragma unroll
    for (int r = 0; r < 8; r++) {
      const int i = q0 + w * 16 + r + 8 * lh;
      float s0 = sf[0][r], s1 = sf[1][r];
      s0 = ((j0 <= i) && (j0 + WIN_ > i)) ? s0 * SCALING_ : -3.0e38f;
      s1 = ((j1 <= i) && (j1 + WIN_ > i)) ? s1 * SCALING_ : -3.0e38f;
      const float mx = redmax16(fmaxf(s0, s1));
      const float mnew = fmaxf(mi_[r], mx);
      const float sc = __expf(mi_[r] - mnew);
      mi_[r] = mnew;
      const float p0 = __expf(s0 - mnew);
      const float p1 = __expf(s1 - mnew);
      sf[0][r] = p0;
      sf[1][r] = p1;
      li_[r] = li_[r] * sc + redsum16(p0 + p1);
#pragma unroll
      for (int f = 0; f < 16; f++) o[f][r] *= sc;
    }

    // ---- P: C-layout -> A-layout via per-wave LDS transpose buffer ----
    const int pb = w * 1280;
#pragma unroll
    for (int ni = 0; ni < 2; ni++)
#pragma unroll
      for (int r = 0; r < 8; r++) {
        const int m = r + 8 * lh;
        *(bf16_t*)(ldsP + pb + m * 80 + (ni * 16 + ln) * 2) = (bf16_t)sf[ni][r];
      }
    v16bf ap = make_frag(ldsP + pb + ln * 80 + lh * 16,
                         ldsP + pb + ln * 80 + 32 + lh * 16);

    // ---- O += P @ V ----
#pragma unroll
    for (int f = 0; f < 16; f++) {
      const int d = f * 16 + ln;
      v16bf bv = make_frag(ldsV + d * 80 + lh * 16,
                           ldsV + d * 80 + 32 + lh * 16);
      o[f] = wmma_bf16(ap, bv, o[f]);
    }
  }

  // ---- epilogue: O / l -> bf16 attn output [B*S, 8*256] ----
#pragma unroll
  for (int r = 0; r < 8; r++) {
    const float inv = 1.0f / li_[r];
    const int srow = q0 + w * 16 + r + 8 * lh;
    bf16_t* op = O + qbase + (size_t)srow * 2048;
#pragma unroll
    for (int f = 0; f < 16; f++)
      op[f * 16 + ln] = (bf16_t)(o[f][r] * inv);
  }
}

// ---------------------------------------------------------------------------
extern "C" void kernel_launch(void* const* d_in, const int* in_sizes, int n_in,
                              void* d_out, int out_size, void* d_ws, size_t ws_size,
                              hipStream_t stream) {
  (void)in_sizes; (void)n_in; (void)out_size; (void)ws_size;
  const float* x  = (const float*)d_in[0];
  const float* wq = (const float*)d_in[1];
  const float* wk = (const float*)d_in[2];
  const float* wv = (const float*)d_in[3];
  const float* wo = (const float*)d_in[4];
  const float* qn = (const float*)d_in[5];
  const float* kn = (const float*)d_in[6];
  float* out = (float*)d_out;

  char* ws = (char*)d_ws;
  bf16_t* Qb = (bf16_t*)(ws);                       // 8192*2048 bf16 = 32 MB
  bf16_t* Kb = (bf16_t*)(ws + (size_t)33554432);    // 8192*1024 bf16 = 16 MB
  bf16_t* Vb = (bf16_t*)(ws + (size_t)50331648);    // 16 MB
  bf16_t* Ob = (bf16_t*)(ws + (size_t)67108864);    // 32 MB

  gemm_f32_bf16<<<dim3(16, 64), 256, 0, stream>>>(x, wq, Qb, 2048, 2048);
  gemm_f32_bf16<<<dim3(8, 64), 256, 0, stream>>>(x, wk, Kb, 1024, 2048);
  gemm_f32_bf16<<<dim3(8, 64), 256, 0, stream>>>(x, wv, Vb, 1024, 2048);
  rmsrope_kernel<<<8192, 256, 0, stream>>>(Qb, qn, 8);
  rmsrope_kernel<<<4096, 256, 0, stream>>>(Kb, kn, 4);
  attn_kernel<<<dim3(64, 8, 2), 128, 0, stream>>>(Qb, Kb, Vb, Ob);
  gemm_bf16_f32<<<dim3(16, 64), 256, 0, stream>>>(Ob, wo, out, 2048, 2048);
}
